// LookupAlignConvolution2d_32186484916363
// MI455X (gfx1250) — matrix operand order repaired and checked by reading, not compiled, not run on hardware
//
#include <hip/hip_runtime.h>
#include <stdint.h>

// ---------------- CDNA5 vector types ----------------
typedef __bf16 bf16_t;
typedef bf16_t v16bf __attribute__((ext_vector_type(16)));
typedef bf16_t v8bf  __attribute__((ext_vector_type(8)));
typedef float  v8f   __attribute__((ext_vector_type(8)));

// ---------------- problem constants ----------------
#define NB   32
#define H    64
#define W    64
#define C    256
#define OC   256
#define OH   62
#define OW   62
#define KDIM (C * 9)          // 2304
#define MTOT (NB * OH * OW)   // 123008 = 961 * 128
#define XTOT (NB * H * W * C) // 33,554,432

// ---------------- tiling (main path): 128 x 256 tile, 8 waves of 64x64 ----
#define BM  128
#define BN  256
#define BK  32
#define KSTEPS 72            // KDIM / BK
#define LDA 48               // padded row stride (bf16 elems): 96 B
#define LDB 48

#define WP_BYTES  ((size_t)OC * KDIM * 2)        // 1,179,648
#define XBF_BYTES ((size_t)XTOT * 2)             // 67,108,864

__device__ __forceinline__ uint32_t f2bf(float f) {
  uint32_t u = __float_as_uint(f);
  return (u + 0x7FFFu + ((u >> 16) & 1u)) >> 16;   // RNE
}
__device__ __forceinline__ uint32_t pack2(float a, float b) {
  return f2bf(a) | (f2bf(b) << 16);
}

// ---- prepack: masked bf16 weights, [co][k] layout ----
__global__ void prep_weights_kernel(const float* __restrict__ w,
                                    uint16_t* __restrict__ wp) {
  int e = blockIdx.x * 256 + threadIdx.x;
  if (e >= OC * KDIM) return;
  int co   = e / KDIM;
  int k    = e - co * KDIM;
  int khkw = k >> 8;
  int ci   = k & 255;
  int kh = khkw / 3, kw = khkw - kh * 3;
  float v = w[(((co * C) + ci) * 3 + kh) * 3 + kw];   // OIHW
  if (fabsf(v) < 0.01f) v = 0.0f;
  wp[e] = (uint16_t)f2bf(v);
}

// ---- prepack: x fp32 -> bf16, same NHWC flat layout ----
__global__ void prep_x_kernel(const float* __restrict__ x,
                              uint16_t* __restrict__ xb) {
  size_t i = ((size_t)blockIdx.x * 256 + threadIdx.x) * 4;   // XTOT % 4 == 0
  float4 f = *(const float4*)(x + i);
  uint2 p;
  p.x = pack2(f.x, f.y);
  p.y = pack2(f.z, f.w);
  *(uint2*)(xb + i) = p;
}

// async b128 copies; INST_OFFSET advances LDS and global addresses together
#define ASYNC_CP32B(ldsoff, voff, sbase)                                    \
  do {                                                                      \
    asm volatile("global_load_async_to_lds_b128 %0, %1, %2"                 \
                 :: "v"(ldsoff), "v"(voff), "s"(sbase) : "memory");         \
    asm volatile("global_load_async_to_lds_b128 %0, %1, %2 offset:16"       \
                 :: "v"(ldsoff), "v"(voff), "s"(sbase) : "memory");         \
  } while (0)
#define ASYNC_CP64B(ldsoff, voff, sbase)                                    \
  do {                                                                      \
    asm volatile("global_load_async_to_lds_b128 %0, %1, %2"                 \
                 :: "v"(ldsoff), "v"(voff), "s"(sbase) : "memory");         \
    asm volatile("global_load_async_to_lds_b128 %0, %1, %2 offset:16"       \
                 :: "v"(ldsoff), "v"(voff), "s"(sbase) : "memory");         \
    asm volatile("global_load_async_to_lds_b128 %0, %1, %2 offset:32"       \
                 :: "v"(ldsoff), "v"(voff), "s"(sbase) : "memory");         \
    asm volatile("global_load_async_to_lds_b128 %0, %1, %2 offset:48"       \
                 :: "v"(ldsoff), "v"(voff), "s"(sbase) : "memory");         \
  } while (0)

// =====================  main path: bf16-prepacked + async DMA  ==============
__launch_bounds__(256)
__global__ void conv_wmma_async_kernel(const uint16_t* __restrict__ xbf,
                                       const uint16_t* __restrict__ wp,
                                       const float* __restrict__ bias,
                                       float* __restrict__ out) {
  __shared__ uint16_t ldsA[2][BM * LDA];   // 2 x 12 KB
  __shared__ uint16_t ldsB[2][BN * LDB];   // 2 x 24 KB

  const int tid    = threadIdx.x;
  const int tile_m = blockIdx.x;

  // ---- A staging: row = tid/2, 16 bf16 (32 B) per thread ----
  const int arow  = tid >> 1;
  const int ahalf = (tid & 1) << 4;
  int mg  = tile_m * BM + arow;
  int nb  = mg / (OH * OW);
  int rem = mg - nb * (OH * OW);
  int oh  = rem / OW;
  int ow  = rem - oh * OW;
  const uint32_t a_elem_base = (uint32_t)(((nb * H + oh) * W + ow) * C) + ahalf;
  // ---- B staging: one output channel (32 bf16 = 64 B) per thread ----
  const uint32_t b_elem_base = (uint32_t)(tid * KDIM);

  uint32_t ldsAoff[2], ldsBoff[2];
  ldsAoff[0] = (uint32_t)(uintptr_t)&ldsA[0][arow * LDA + ahalf];
  ldsAoff[1] = (uint32_t)(uintptr_t)&ldsA[1][arow * LDA + ahalf];
  ldsBoff[0] = (uint32_t)(uintptr_t)&ldsB[0][tid * LDB];
  ldsBoff[1] = (uint32_t)(uintptr_t)&ldsB[1][tid * LDB];

  // ---- wave/lane mapping: 8 waves, 2 (M) x 4 (N), 64x64 each ----
  const int wave = tid >> 5;
  const int lane = tid & 31;
  const int wm   = wave & 1;
  const int wn   = wave >> 1;
  const int l15  = lane & 15;
  const int lhi  = lane >> 4;
  const int akbase = lhi << 3;
  const int bkbase = lhi << 3;

  v8f acc[4][4];
#pragma unroll
  for (int i = 0; i < 4; ++i)
#pragma unroll
    for (int j = 0; j < 4; ++j) acc[i][j] = (v8f)0.0f;

  // ---- prologue: async-stage step 0 into buffer 0 ----
  {
    const uint32_t aoff = a_elem_base * 2;
    const uint32_t boff = b_elem_base * 2;
    ASYNC_CP32B(ldsAoff[0], aoff, xbf);
    ASYNC_CP64B(ldsBoff[0], boff, wp);
  }

#pragma unroll 1
  for (int step = 0; step < KSTEPS; ++step) {
    const int cur = step & 1;

    if (step + 1 < KSTEPS) {
      const int ns  = step + 1;
      const int tap = ns >> 3;
      const int khv = tap / 3;
      const int kwv = tap - khv * 3;
      const uint32_t koffA = (uint32_t)(((khv * W + kwv) << 8) + ((ns & 7) << 5));
      const uint32_t koffB = (uint32_t)((tap << 8) + ((ns & 7) << 5));
      const uint32_t aoff  = (a_elem_base + koffA) * 2;
      const uint32_t boff  = (b_elem_base + koffB) * 2;
      ASYNC_CP32B(ldsAoff[1 - cur], aoff, xbf);
      ASYNC_CP64B(ldsBoff[1 - cur], boff, wp);
      asm volatile("s_wait_asynccnt 0x6" ::: "memory");  // current step's 6 done
    } else {
      asm volatile("s_wait_asynccnt 0x0" ::: "memory");
    }
    __syncthreads();   // publish current buffer to all waves

    // ---- fragments per ISA 16-bit A/B VGPR layout ----
    v16bf afrag[4], bfrag[4];
#pragma unroll
    for (int i = 0; i < 4; ++i) {
      const uint16_t* ap = &ldsA[cur][(wm * 64 + i * 16 + l15) * LDA + akbase];
      v8bf lo = *(const v8bf*)(ap);        // K = akbase .. +7
      v8bf hi = *(const v8bf*)(ap + 16);   // K = akbase+16 .. +23
      afrag[i] = __builtin_shufflevector(lo, hi, 0, 1, 2, 3, 4, 5, 6, 7,
                                         8, 9, 10, 11, 12, 13, 14, 15);
    }
#pragma unroll
    for (int j = 0; j < 4; ++j) {
      const uint16_t* bp = &ldsB[cur][(wn * 64 + j * 16 + l15) * LDB + bkbase];
      v8bf lo = *(const v8bf*)(bp);
      v8bf hi = *(const v8bf*)(bp + 16);
      bfrag[j] = __builtin_shufflevector(lo, hi, 0, 1, 2, 3, 4, 5, 6, 7,
                                         8, 9, 10, 11, 12, 13, 14, 15);
    }

    // ---- 16x v_wmma_f32_16x16x32_bf16 per wave per k-step ----
#pragma unroll
    for (int i = 0; i < 4; ++i)
#pragma unroll
      for (int j = 0; j < 4; ++j)
        acc[i][j] = __builtin_amdgcn_wmma_f32_16x16x32_bf16(
            false, afrag[i], false, bfrag[j],
            (short)0, acc[i][j], false, false);

    __syncthreads();   // all reads of buf[cur] done before step+2 re-stages it
  }

  // ---- epilogue: fused bias add, f32 stores (NHWC flat = m*256 + co) ----
#pragma unroll
  for (int j = 0; j < 4; ++j) {
    const int col = wn * 64 + j * 16 + l15;
    const float bv = bias[col];
#pragma unroll
    for (int i = 0; i < 4; ++i) {
      const int mg0 = tile_m * BM + wm * 64 + i * 16 + (lhi << 3);
      float* op = out + (size_t)mg0 * OC + col;
#pragma unroll
      for (int v = 0; v < 8; ++v)
        op[(size_t)v * OC] = acc[i][j][v] + bv;
    }
  }
}

// ==============  fallback path: convert-on-the-fly (small d_ws)  ============
__launch_bounds__(256)
__global__ void conv_wmma_fallback_kernel(const float* __restrict__ x,
                                          const uint16_t* __restrict__ wp,
                                          const float* __restrict__ bias,
                                          float* __restrict__ out) {
  __shared__ uint16_t ldsA[128 * LDA];
  __shared__ uint16_t ldsB[128 * LDB];

  const int tid    = threadIdx.x;
  const int tile_n = blockIdx.x & 1;
  const int tile_m = blockIdx.x >> 1;
  const int co0    = tile_n * 128;

  const int arow  = tid >> 1;
  const int ahalf = (tid & 1) << 4;
  int mg  = tile_m * 128 + arow;
  int nb  = mg / (OH * OW);
  int rem = mg - nb * (OH * OW);
  int oh  = rem / OW;
  int ow  = rem - oh * OW;
  const float* xbase = x + (size_t)(((nb * H + oh) * W + ow)) * C;
  const uint16_t* wbase = wp + (size_t)(co0 + arow) * KDIM + ahalf;

  const int wave = tid >> 5;
  const int lane = tid & 31;
  const int wm   = wave & 1;
  const int wn   = wave >> 1;
  const int l15  = lane & 15;
  const int lhi  = lane >> 4;
  const int akbase = lhi << 3;
  const int bkbase = lhi << 3;

  v8f acc[4][2];
#pragma unroll
  for (int i = 0; i < 4; ++i)
#pragma unroll
    for (int j = 0; j < 2; ++j) acc[i][j] = (v8f)0.0f;

#pragma unroll 1
  for (int step = 0; step < KSTEPS; ++step) {
    const int tap = step >> 3;
    const int khv = tap / 3;
    const int kwv = tap - khv * 3;
    const int ci0 = (step & 7) << 5;
    __syncthreads();
    {
      const float* src = xbase + ((khv * W + kwv) * C) + ci0 + ahalf;
      float4 f0 = *(const float4*)(src + 0);
      float4 f1 = *(const float4*)(src + 4);
      float4 f2 = *(const float4*)(src + 8);
      float4 f3 = *(const float4*)(src + 12);
      uint4 p0, p1;
      p0.x = pack2(f0.x, f0.y); p0.y = pack2(f0.z, f0.w);
      p0.z = pack2(f1.x, f1.y); p0.w = pack2(f1.z, f1.w);
      p1.x = pack2(f2.x, f2.y); p1.y = pack2(f2.z, f2.w);
      p1.z = pack2(f3.x, f3.y); p1.w = pack2(f3.z, f3.w);
      uint4* dst = (uint4*)&ldsA[arow * LDA + ahalf];
      dst[0] = p0; dst[1] = p1;
    }
    {
      const uint16_t* src = wbase + (tap << 8) + ci0;
      uint4 b0 = *(const uint4*)(src + 0);
      uint4 b1 = *(const uint4*)(src + 8);
      uint4* dst = (uint4*)&ldsB[arow * LDB + ahalf];
      dst[0] = b0; dst[1] = b1;
    }
    __syncthreads();

    v16bf afrag[4], bfrag[2];
#pragma unroll
    for (int i = 0; i < 4; ++i) {
      const uint16_t* ap = &ldsA[(wm * 64 + i * 16 + l15) * LDA + akbase];
      v8bf lo = *(const v8bf*)(ap);
      v8bf hi = *(const v8bf*)(ap + 16);
      afrag[i] = __builtin_shufflevector(lo, hi, 0, 1, 2, 3, 4, 5, 6, 7,
                                         8, 9, 10, 11, 12, 13, 14, 15);
    }
#pragma unroll
    for (int j = 0; j < 2; ++j) {
      const uint16_t* bp = &ldsB[(wn * 32 + j * 16 + l15) * LDB + bkbase];
      v8bf lo = *(const v8bf*)(bp);
      v8bf hi = *(const v8bf*)(bp + 16);
      bfrag[j] = __builtin_shufflevector(lo, hi, 0, 1, 2, 3, 4, 5, 6, 7,
                                         8, 9, 10, 11, 12, 13, 14, 15);
    }
#pragma unroll
    for (int i = 0; i < 4; ++i)
#pragma unroll
      for (int j = 0; j < 2; ++j)
        acc[i][j] = __builtin_amdgcn_wmma_f32_16x16x32_bf16(
            false, afrag[i], false, bfrag[j],
            (short)0, acc[i][j], false, false);
  }

#pragma unroll
  for (int j = 0; j < 2; ++j) {
    const int col = co0 + wn * 32 + j * 16 + l15;
    const float bv = bias[col];
#pragma unroll
    for (int i = 0; i < 4; ++i) {
      const int mg0 = tile_m * 128 + wm * 64 + i * 16 + (lhi << 3);
      float* op = out + (size_t)mg0 * OC + col;
#pragma unroll
      for (int v = 0; v < 8; ++v)
        op[(size_t)v * OC] = acc[i][j][v] + bv;
    }
  }
}

extern "C" void kernel_launch(void* const* d_in, const int* in_sizes, int n_in,
                              void* d_out, int out_size, void* d_ws, size_t ws_size,
                              hipStream_t stream) {
  const float* x    = (const float*)d_in[0];   // (32,64,64,256) NHWC fp32
  const float* w    = (const float*)d_in[1];   // (256,256,3,3)  OIHW fp32
  const float* bias = (const float*)d_in[2];   // (256,)
  float* out        = (float*)d_out;           // (32,62,62,256) fp32

  uint16_t* wp = (uint16_t*)d_ws;
  prep_weights_kernel<<<(OC * KDIM + 255) / 256, 256, 0, stream>>>(w, wp);

  if (ws_size >= WP_BYTES + XBF_BYTES) {
    uint16_t* xbf = (uint16_t*)((char*)d_ws + WP_BYTES);
    prep_x_kernel<<<XTOT / 4 / 256, 256, 0, stream>>>(x, xbf);
    conv_wmma_async_kernel<<<MTOT / BM, 256, 0, stream>>>(xbf, wp, bias, out);
  } else {
    conv_wmma_fallback_kernel<<<(MTOT / 128) * 2, 256, 0, stream>>>(x, wp, bias, out);
  }
}
